// ScaledDotProductAttention_63282048139553
// MI455X (gfx1250) — matrix operand order repaired and checked by reading, not compiled, not run on hardware
//
#include <hip/hip_runtime.h>

typedef __attribute__((ext_vector_type(16))) __bf16 v16bf;
typedef __attribute__((ext_vector_type(8)))  __bf16 v8bf;
typedef __attribute__((ext_vector_type(4)))  __bf16 v4bf;
typedef __attribute__((ext_vector_type(8)))  float  v8f;

#define Bb 4
#define Hh 16
#define Ss 2048
#define Dd 64

#define KP 72            // bf16 per LDS K row (64 + 8 pad)
#define VP 40            // bf16 per LDS V^T row (32 + 8 pad)
#define PP 40            // bf16 per LDS P row
#define KT (32 * KP)
#define VT (64 * VP)

__device__ __forceinline__ v16bf ld_frag16(const __bf16* p) {
    v8bf a = *(const v8bf*)p;
    v8bf b = *(const v8bf*)(p + 8);
    return __builtin_shufflevector(a, b, 0,1,2,3,4,5,6,7,8,9,10,11,12,13,14,15);
}

__global__ __launch_bounds__(256)
void fattn_fwd_kernel(const float* __restrict__ q,
                      const float* __restrict__ k,
                      const float* __restrict__ v,
                      const int*   __restrict__ mask,
                      float* __restrict__ outO,   // [B,H,S,D]
                      float* __restrict__ outA)   // [B,H,S,S]
{
    __shared__ __bf16 sK [2][KT];      // K tiles, bf16 row-major, double buffered
    __shared__ __bf16 sVt[2][VT];      // V tiles, bf16 transposed, double buffered
    __shared__ __bf16 sP [8][16 * PP]; // per-wave P tile (16 x 32)

    const int tid  = threadIdx.x;
    const int wave = tid >> 5;
    const int lane = tid & 31;
    const int hi   = lane >> 4;
    const int ln   = lane & 15;

    const int bh    = blockIdx.x >> 4;   // b*H + h
    const int qblk  = blockIdx.x & 15;
    const int b     = bh >> 4;           // /H (H==16)
    const int qrow0 = qblk * 128 + wave * 16;

    const float* qbase = q + (size_t)bh * Ss * Dd;
    const float* kbase = k + (size_t)bh * Ss * Dd;
    const float* vbase = v + (size_t)bh * Ss * Dd;
    const int*   mrow  = mask + b * Ss;

    // cooperative staging coordinates: 256 threads cover 32 rows x 64 cols (8 f32/thread)
    const int srow = tid >> 3;
    const int scb  = (tid & 7) * 8;

    // ---- Q A-fragments (two 32-deep k-steps), contiguous k = 16*hi + i order (same as B) ----
    v16bf aq[2];
    {
        const float* qr = qbase + (size_t)(qrow0 + ln) * Dd;
#pragma unroll
        for (int s = 0; s < 2; ++s) {
            const int kb = 32 * s + 16 * hi;
#pragma unroll
            for (int i = 0; i < 8; ++i) {
                aq[s][i]     = (__bf16)qr[kb + i];
                aq[s][8 + i] = (__bf16)qr[kb + 8 + i];
            }
        }
    }

    float Sl[8];
#pragma unroll
    for (int r = 0; r < 8; ++r) Sl[r] = 0.0f;

    // ================= PASS 1: sum of exp(score)*mask (no max needed; see analysis) ====
    float4 kf0, kf1;
    {   // prefetch tile 0
        const size_t g = (size_t)srow * Dd + scb;
        kf0 = *(const float4*)&kbase[g];
        kf1 = *(const float4*)&kbase[g + 4];
    }
    for (int jp = 0; jp < Ss / 32; ++jp) {
        const int buf = jp & 1;
        {   // commit prefetched K tile to LDS (bf16)
            v4bf k0 = { (__bf16)kf0.x, (__bf16)kf0.y, (__bf16)kf0.z, (__bf16)kf0.w };
            v4bf k1 = { (__bf16)kf1.x, (__bf16)kf1.y, (__bf16)kf1.z, (__bf16)kf1.w };
            *(v4bf*)&sK[buf][srow * KP + scb]     = k0;
            *(v4bf*)&sK[buf][srow * KP + scb + 4] = k1;
        }
        __syncthreads();
        if (jp + 1 < Ss / 32) {   // prefetch next tile while computing this one
            const size_t g = (size_t)((jp + 1) * 32 + srow) * Dd + scb;
            kf0 = *(const float4*)&kbase[g];
            kf1 = *(const float4*)&kbase[g + 4];
        }
#pragma unroll
        for (int jj = 0; jj < 2; ++jj) {
            v8f acc = {};
#pragma unroll
            for (int s = 0; s < 2; ++s) {
                v16bf bk = ld_frag16(&sK[buf][(jj * 16 + ln) * KP + 32 * s + 16 * hi]);
                acc = __builtin_amdgcn_wmma_f32_16x16x32_bf16(
                          false, aq[s], false, bk, (short)0, acc, false, false);
            }
            const float mkf = (float)mrow[jp * 32 + jj * 16 + ln];  // 0.0 or 1.0
#pragma unroll
            for (int r = 0; r < 8; ++r)
                Sl[r] = fmaf(__expf(acc[r]), mkf, Sl[r]);
        }
    }

    // ---- single cross-lane sum merge (4-stage butterfly within 16-lane halves) ----
    float invS[8];
#pragma unroll
    for (int r = 0; r < 8; ++r) {
        float s = Sl[r];
        s += __shfl_xor(s, 1, 32);
        s += __shfl_xor(s, 2, 32);
        s += __shfl_xor(s, 4, 32);
        s += __shfl_xor(s, 8, 32);
        invS[r] = 1.0f / s;
    }

    // ========= PASS 2: recompute scores, write attn, accumulate P.V =========
    v8f oacc[4] = {};
    float* attn = outA + (size_t)bh * Ss * Ss;

    float4 vf0, vf1;
    {   // prefetch tile 0 (K + V)
        const size_t g = (size_t)srow * Dd + scb;
        kf0 = *(const float4*)&kbase[g];
        kf1 = *(const float4*)&kbase[g + 4];
        vf0 = *(const float4*)&vbase[g];
        vf1 = *(const float4*)&vbase[g + 4];
    }
    for (int jp = 0; jp < Ss / 32; ++jp) {
        const int buf = jp & 1;
        {   // commit prefetched K (row-major) + V (transposed) tiles as bf16
            v4bf k0 = { (__bf16)kf0.x, (__bf16)kf0.y, (__bf16)kf0.z, (__bf16)kf0.w };
            v4bf k1 = { (__bf16)kf1.x, (__bf16)kf1.y, (__bf16)kf1.z, (__bf16)kf1.w };
            *(v4bf*)&sK[buf][srow * KP + scb]     = k0;
            *(v4bf*)&sK[buf][srow * KP + scb + 4] = k1;
            __bf16* vt = &sVt[buf][0];
            vt[(scb + 0) * VP + srow] = (__bf16)vf0.x;
            vt[(scb + 1) * VP + srow] = (__bf16)vf0.y;
            vt[(scb + 2) * VP + srow] = (__bf16)vf0.z;
            vt[(scb + 3) * VP + srow] = (__bf16)vf0.w;
            vt[(scb + 4) * VP + srow] = (__bf16)vf1.x;
            vt[(scb + 5) * VP + srow] = (__bf16)vf1.y;
            vt[(scb + 6) * VP + srow] = (__bf16)vf1.z;
            vt[(scb + 7) * VP + srow] = (__bf16)vf1.w;
        }
        __syncthreads();
        if (jp + 1 < Ss / 32) {
            const size_t g = (size_t)((jp + 1) * 32 + srow) * Dd + scb;
            kf0 = *(const float4*)&kbase[g];
            kf1 = *(const float4*)&kbase[g + 4];
            vf0 = *(const float4*)&vbase[g];
            vf1 = *(const float4*)&vbase[g + 4];
        }

#pragma unroll
        for (int jj = 0; jj < 2; ++jj) {
            v8f acc = {};
#pragma unroll
            for (int s = 0; s < 2; ++s) {
                v16bf bk = ld_frag16(&sK[buf][(jj * 16 + ln) * KP + 32 * s + 16 * hi]);
                acc = __builtin_amdgcn_wmma_f32_16x16x32_bf16(
                          false, aq[s], false, bk, (short)0, acc, false, false);
            }
            const int col = jp * 32 + jj * 16 + ln;
            const float mkf = (float)mrow[col];   // 0.0 or 1.0
#pragma unroll
            for (int r = 0; r < 8; ++r) {
                float p = __expf(acc[r]) * (mkf * invS[r]);   // exact 0 for masked cols
                const int row = qrow0 + r + 8 * hi;
                __builtin_nontemporal_store(p, &attn[(size_t)row * Ss + col]);
                sP[wave][(r + 8 * hi) * PP + jj * 16 + ln] = (__bf16)p;
            }
        }

        __syncthreads();   // sP C-layout writes -> A-layout reads; also fences sVt readers

        v16bf pa;
        {
            const __bf16* pr = &sP[wave][ln * PP + 16 * hi];
            v8bf a = *(const v8bf*)pr;
            v8bf c = *(const v8bf*)(pr + 8);
            pa = __builtin_shufflevector(a, c, 0,1,2,3,4,5,6,7,8,9,10,11,12,13,14,15);
        }
#pragma unroll
        for (int t = 0; t < 4; ++t) {
            v16bf bv = ld_frag16(&sVt[buf][(16 * t + ln) * VP + 16 * hi]);
            oacc[t] = __builtin_amdgcn_wmma_f32_16x16x32_bf16(
                          false, pa, false, bv, (short)0, oacc[t], false, false);
        }
    }

    // ---- write output [B,H,S,D] (write-once -> non-temporal) ----
#pragma unroll
    for (int t = 0; t < 4; ++t)
#pragma unroll
        for (int r = 0; r < 8; ++r)
            __builtin_nontemporal_store(
                oacc[t][r],
                &outO[((size_t)bh * Ss + qrow0 + r + 8 * hi) * Dd + t * 16 + ln]);
}

extern "C" void kernel_launch(void* const* d_in, const int* in_sizes, int n_in,
                              void* d_out, int out_size, void* d_ws, size_t ws_size,
                              hipStream_t stream) {
    (void)in_sizes; (void)n_in; (void)out_size; (void)d_ws; (void)ws_size;
    const float* q    = (const float*)d_in[0];
    const float* k    = (const float*)d_in[1];
    const float* v    = (const float*)d_in[2];
    const int*   mask = (const int*)d_in[3];

    float* outO = (float*)d_out;                          // B*H*S*D
    float* outA = outO + (size_t)Bb * Hh * Ss * Dd;       // B*H*S*S

    fattn_fwd_kernel<<<dim3(64 * 16), dim3(256), 0, stream>>>(q, k, v, mask, outO, outA);
}
